// attn_17575006175898
// MI455X (gfx1250) — matrix operand order repaired and checked by reading, not compiled, not run on hardware
//
#include <hip/hip_runtime.h>

// ---------------------------------------------------------------------------
// Fused XCiT-style window attention for MI455X (gfx1250, wave32, WMMA).
//
// One workgroup per 8x8 window (4096 windows), 256 threads = 8 wave32.
// All GEMMs: v_wmma_f32_16x16x32_f16 (f16 in, f32 accumulate).
// Every WMMA operand is stored in LDS in the major order its consumer needs,
// so each fragment load is 2x ds_load_b128 per lane (no scalar gathers).
// Full fusion: HBM traffic ~= read x (100MB) + write out (100MB) -> memory
// bound at the 23.3 TB/s floor (~9us ideal); 28.7 GFLOP rides on WMMA.
// ---------------------------------------------------------------------------

typedef __attribute__((ext_vector_type(16))) _Float16 v16h;
typedef __attribute__((ext_vector_type(8)))  _Float16 v8h;
typedef __attribute__((ext_vector_type(8)))  float    v8f;

__device__ __forceinline__ v8f wmma32f16(v16h a, v16h b, v8f c) {
  // 8 args: (neg_a, A, neg_b, B, c_mod, C, reuse_a, reuse_b)
  return __builtin_amdgcn_wmma_f32_16x16x32_f16(false, a, false, b, (short)0, c,
                                                false, false);
}

__device__ __forceinline__ v8h zero8h() {
  v8h z;
#pragma unroll
  for (int e = 0; e < 8; ++e) z[e] = (_Float16)0.0f;
  return z;
}
__device__ __forceinline__ v16h cat(v8h lo, v8h hi) {
  return __builtin_shufflevector(lo, hi, 0, 1, 2, 3, 4, 5, 6, 7,
                                 8, 9, 10, 11, 12, 13, 14, 15);
}

// --- Contiguous fragment loaders (CDNA5 wave32 WMMA VGPR layouts) ----------
// A (16x32): lanes 0-15 row m=lane hold K={0..7,16..23}; lanes 16-31 row
// m=lane-16 hold K={8..15,24..31}.  B (32x16): col n=lane&15, same K split as A.
// C/D: VGPR r -> M=r+8*(lane>=16), N=lane&15.

// A from row-major MxK storage, full K=32 tile. p,ld must be 16B aligned.
__device__ __forceinline__ v16h load_a_rm32(const _Float16* p, int ld) {
  const int lane = threadIdx.x & 31;
  const _Float16* r = p + (lane & 15) * ld + ((lane & 16) ? 8 : 0);
  return cat(*(const v8h*)r, *(const v8h*)(r + 16));
}

// A from row-major MxK storage, K=16 real (cols 16..31 zero).
__device__ __forceinline__ v16h load_a_rm16(const _Float16* p, int ld) {
  const int lane = threadIdx.x & 31;
  const _Float16* r = p + (lane & 15) * ld + ((lane & 16) ? 8 : 0);
  return cat(*(const v8h*)r, zero8h());  // elems 8..15 map to K>=16 -> 0
}

// B from transposed (NxK row-major) storage: B[k][n] = p[n*ld + k], K=32.
__device__ __forceinline__ v16h load_b_tr32(const _Float16* p, int ld) {
  const int lane = threadIdx.x & 31;
  const _Float16* r = p + (lane & 15) * ld + ((lane & 16) ? 8 : 0);
  return cat(*(const v8h*)r, *(const v8h*)(r + 16));
}

// B from transposed storage, K=16 real (rows 16..31 zero).
__device__ __forceinline__ v16h load_b_tr16(const _Float16* p, int ld) {
  const int lane = threadIdx.x & 31;
  v8h lo = zero8h(), hi = zero8h();
  {  // elems 8..15 cover K>=16 -> zero; lanes 16-31 elems 0..7 are K=8..15
    const _Float16* r = p + (lane & 15) * ld + ((lane & 16) ? 8 : 0);
    lo = *(const v8h*)r;
  }
  return cat(lo, hi);
}

// ---------------------------------------------------------------------------

namespace cfg {
constexpr int HH = 512, WW = 512;
constexpr int CH = 96, CH3 = 288;
constexpr int HEADS = 6;
constexpr int NPIX = 64;
// LDS arena, phase-aliased (total 165,888 B of the 320KB/WGP budget):
//  A [0,67584):       wqk 288x96 (55296) + xlT 64x96 (12288)
//                      -> aspT 6x[64][64] (49152)          [phase 6-7]
//                      -> pwl 96x96 (18432) + o2sT 64x96 (12288)  [phase 8]
//  B [67584,104448):  raw 288x64 (36864)
//                      -> achf/ach (6144) + o1 6x[16][64] (12288) [phase 4+]
//  C [104448,165888): proc_qk 192x64 (24576) + qT,kT,vT 6x[64][16] (3x12288)
constexpr int SMEM_BYTES = 165888;
constexpr int OFF_XLT  = 55296;
constexpr int OFF_O2ST = 18432;
constexpr int OFF_B    = 67584;
constexpr int OFF_O1   = 67584 + 6144;
constexpr int OFF_C    = 104448;
constexpr int OFF_QT   = 104448 + 24576;
constexpr int OFF_KT   = OFF_QT + 12288;
constexpr int OFF_VT   = OFF_KT + 12288;
}  // namespace cfg

__global__ void prep_weights_f16(const float* __restrict__ qw,
                                 const float* __restrict__ pw,
                                 _Float16* __restrict__ qw_h,
                                 _Float16* __restrict__ pw_h) {
  int i = blockIdx.x * blockDim.x + threadIdx.x;
  if (i < cfg::CH3 * cfg::CH) qw_h[i] = (_Float16)qw[i];
  if (i < cfg::CH  * cfg::CH) pw_h[i] = (_Float16)pw[i];
}

__global__ __launch_bounds__(256) void fused_window_attn(
    const float* __restrict__ x,          // (96,512,512)
    const _Float16* __restrict__ qw_h,    // (288,96) f16
    const float* __restrict__ qkv_b,      // (288,)
    const float* __restrict__ dw_w,       // (288,1,3,3)
    const float* __restrict__ dw_b,       // (288,)
    const _Float16* __restrict__ pw_h,    // (96,96) f16
    const float* __restrict__ proj_b,     // (96,)
    const float* __restrict__ temperature,// (6,)
    const float* __restrict__ rel_tab,    // (225,6)
    float* __restrict__ out)              // (96,512,512)
{
  using namespace cfg;
  __shared__ __align__(16) char smem[SMEM_BYTES];
  _Float16* wqk   = (_Float16*)(smem);               // [288][96] A, row-major
  _Float16* xlT   = (_Float16*)(smem + OFF_XLT);     // [64][96]  B via tr
  _Float16* aspT  = (_Float16*)(smem);               // [6][64][64] (reuse A)
  _Float16* pwl   = (_Float16*)(smem);               // [96][96]    (reuse A)
  _Float16* o2sT  = (_Float16*)(smem + OFF_O2ST);    // [64][96]    (reuse A)
  _Float16* raw   = (_Float16*)(smem + OFF_B);       // [288][64]
  float*    achf  = (float*)   (smem + OFF_B);       // [6][16][16] f32 stage
  _Float16* ach   = (_Float16*)(smem + OFF_B);       // [6][16][32] same bytes
  _Float16* o1    = (_Float16*)(smem + OFF_O1);      // [6][16][64]
  _Float16* proc  = (_Float16*)(smem + OFF_C);       // [192][64] q rows 0..95,
                                                     //            k rows 96..191
  _Float16* qT    = (_Float16*)(smem + OFF_QT);      // [6][64][16]
  _Float16* kT    = (_Float16*)(smem + OFF_KT);      // [6][64][16]
  _Float16* vT    = (_Float16*)(smem + OFF_VT);      // [6][64][16]

  const int wid  = blockIdx.x;
  const int wy   = wid >> 6;
  const int wx   = wid & 63;
  const int tid  = threadIdx.x;
  const int lane = tid & 31;
  const int wave = tid >> 5;
  const int mrow = (lane & 16) ? 8 : 0;
  const int ncol = lane & 15;

  // ---- Phase 1: stage qkv weights (b128) + x tile (pixel-major f16) ------
  for (int i = tid; i < (CH3 * CH) / 8; i += 256)
    ((v8h*)wqk)[i] = ((const v8h*)qw_h)[i];
  for (int i = tid; i < CH * 8; i += 256) {        // one 8-px image row each
    int c = i >> 3, ty = i & 7;
    const float* xr = x + c * (HH * WW) + (wy * 8 + ty) * WW + wx * 8;
#pragma unroll
    for (int e = 0; e < 8; ++e) xlT[(ty * 8 + e) * CH + c] = (_Float16)xr[e];
  }
  __syncthreads();

  // ---- Phase 2: qkv 1x1 conv GEMM (288x64 = 288x96 @ 96x64), 72 tiles ----
  for (int it = 0; it < 9; ++it) {
    int id = wave * 9 + it;
    int mo = id >> 2, no = id & 3;
    v8f acc = {};
#pragma unroll
    for (int k0 = 0; k0 < CH; k0 += 32)
      acc = wmma32f16(load_a_rm32(wqk + mo * 16 * CH + k0, CH),
                      load_b_tr32(xlT + (no * 16) * CH + k0, CH), acc);
#pragma unroll
    for (int r = 0; r < 8; ++r) {
      int o = mo * 16 + r + mrow;
      int t = no * 16 + ncol;
      raw[o * NPIX + t] = (_Float16)(acc[r] + qkv_b[o]);
    }
  }
  __syncthreads();

  // ---- Phase 3: depthwise 3x3 (window-local SAME) + L2 normalize q,k -----
  // (q *= hd^-0.5 precedes F.normalize in the reference -> cancels.)
  for (int c = tid; c < CH3; c += 256) {
    float wv[9];
#pragma unroll
    for (int j = 0; j < 9; ++j) wv[j] = dw_w[c * 9 + j];
    float rin[64];
    const v8h* rv = (const v8h*)(raw + c * NPIX);
#pragma unroll
    for (int j = 0; j < 8; ++j) {
      v8h t = rv[j];
#pragma unroll
      for (int e = 0; e < 8; ++e) rin[j * 8 + e] = (float)t[e];
    }
    float bias = dw_b[c];
    float tmp[64];
#pragma unroll
    for (int t = 0; t < 64; ++t) {
      int ty = t >> 3, tx = t & 7;
      float a = bias;
#pragma unroll
      for (int ky = 0; ky < 3; ++ky) {
        int yy = ty + ky - 1;
        if ((unsigned)yy < 8u) {
#pragma unroll
          for (int kx = 0; kx < 3; ++kx) {
            int xx = tx + kx - 1;
            if ((unsigned)xx < 8u) a += wv[ky * 3 + kx] * rin[yy * 8 + xx];
          }
        }
      }
      tmp[t] = a;
    }
    float inv = 1.0f;
    if (c < 192) {  // q,k: L2 norm over the 64 spatial values, eps 1e-12
      float ss = 0.f;
#pragma unroll
      for (int t = 0; t < 64; ++t) ss += tmp[t] * tmp[t];
      inv = 1.0f / fmaxf(sqrtf(ss), 1e-12f);
    }
    // channel-major copy (q,k only), vectorized
    if (c < 192) {
#pragma unroll
      for (int j = 0; j < 8; ++j) {
        v8h o;
#pragma unroll
        for (int e = 0; e < 8; ++e) o[e] = (_Float16)(tmp[j * 8 + e] * inv);
        *(v8h*)(proc + c * NPIX + j * 8) = o;
      }
    }
    // pixel-major (transposed) copy for q,k,v
    const int h = (c % 96) >> 4, cl = c & 15;
    _Float16* tp = (c < 96 ? qT : (c < 192 ? kT : vT)) + h * 1024 + cl;
#pragma unroll
    for (int t = 0; t < 64; ++t) tp[t * 16] = (_Float16)(tmp[t] * inv);
  }
  __syncthreads();

  // ---- Phase 4: channel attn 16x16 = q(16x64) @ k^T, wave h = head h -----
  if (wave < HEADS) {
    const int h = wave;
    v8f acc = {};
#pragma unroll
    for (int k0 = 0; k0 < 64; k0 += 32)
      acc = wmma32f16(load_a_rm32(proc + (h * 16) * NPIX + k0, NPIX),
                      load_b_tr32(proc + (96 + h * 16) * NPIX + k0, NPIX), acc);
    float tmp = temperature[h];
#pragma unroll
    for (int r = 0; r < 8; ++r)
      achf[h * 256 + (r + mrow) * 16 + ncol] = acc[r] * tmp;
  }
  __syncthreads();
  // softmax over last dim (16); rewrite in place as f16 16x32, K zero-padded
  for (int row = tid; row < 96; row += 256) {
    int h = row >> 4, m = row & 15;
    float v[16], mx = -1e30f;
#pragma unroll
    for (int d = 0; d < 16; ++d) { v[d] = achf[h * 256 + m * 16 + d]; mx = fmaxf(mx, v[d]); }
    float s = 0.f;
#pragma unroll
    for (int d = 0; d < 16; ++d) { v[d] = __expf(v[d] - mx); s += v[d]; }
    float inv = 1.0f / s;
#pragma unroll
    for (int j = 0; j < 4; ++j) {
      v8h o;
#pragma unroll
      for (int e = 0; e < 8; ++e) {
        int d = j * 8 + e;
        o[e] = (d < 16) ? (_Float16)(v[d] * inv) : (_Float16)0.0f;
      }
      *(v8h*)(ach + h * 512 + m * 32 + j * 8) = o;
    }
  }
  __syncthreads();

  // ---- Phase 5: out1 = attn_ch(16x16) @ v(16x64), K=16 padded ------------
  if (wave < HEADS) {
    const int h = wave;
    v16h afrag = load_a_rm32(ach + h * 512, 32);   // zero-padded cols 16..31
    for (int j = 0; j < 4; ++j) {
      v8f acc = {};
      acc = wmma32f16(afrag, load_b_tr16(vT + h * 1024 + (j * 16) * 16, 16), acc);
#pragma unroll
      for (int r = 0; r < 8; ++r)
        o1[h * 1024 + (r + mrow) * NPIX + j * 16 + ncol] = (_Float16)acc[r];
    }
  }
  __syncthreads();

  // ---- Phase 6: spatial attn, computed transposed: -----------------------
  // aspT[m][n] = sum_c k[c,m] q[c,n] = attn[n,m];  A=kT row-major, B=qT^T.
  if (wave < HEADS) {
    const int h = wave;
    for (int i = 0; i < 4; ++i)
      for (int j = 0; j < 4; ++j) {
        v8f acc = {};
        acc = wmma32f16(load_a_rm16(kT + h * 1024 + (i * 16) * 16, 16),
                        load_b_tr16(qT + h * 1024 + (j * 16) * 16, 16), acc);
#pragma unroll
        for (int r = 0; r < 8; ++r) {
          int msp = i * 16 + r + mrow;   // original col m
          int nsp = j * 16 + ncol;       // original row n
          int ry = (nsp >> 3) - (msp >> 3) + 7;
          int rx = (nsp & 7) - (msp & 7) + 7;
          float b = rel_tab[(ry * 15 + rx) * HEADS + h];
          aspT[h * 4096 + msp * 64 + nsp] = (_Float16)(acc[r] + b);
        }
      }
  }
  __syncthreads();
  // softmax over m (a column of aspT) for each fixed n
  for (int row = tid; row < 384; row += 256) {
    int h = row >> 6, n = row & 63;
    _Float16* rp = aspT + h * 4096 + n;
    float v[64], mx = -1e30f;
#pragma unroll
    for (int m = 0; m < 64; ++m) { v[m] = (float)rp[m * 64]; mx = fmaxf(mx, v[m]); }
    float s = 0.f;
#pragma unroll
    for (int m = 0; m < 64; ++m) { v[m] = __expf(v[m] - mx); s += v[m]; }
    float inv = 1.0f / s;
#pragma unroll
    for (int m = 0; m < 64; ++m) rp[m * 64] = (_Float16)(v[m] * inv);
  }
  __syncthreads();

  // ---- Phase 7: out2, computed transposed: out2T = aspT @ o1^T -----------
  // out2T[m][c] = sum_n aspT[m][n] * o1[c][n]
  v8f o2[4] = {};
  if (wave < HEADS) {
    const int h = wave;
    for (int i = 0; i < 4; ++i) {
      v8f acc = {};
#pragma unroll
      for (int k0 = 0; k0 < 64; k0 += 32)
        acc = wmma32f16(load_a_rm32(aspT + h * 4096 + (i * 16) * 64 + k0, 64),
                        load_b_tr32(o1 + h * 1024 + k0, NPIX), acc);
      o2[i] = acc;
    }
  }
  __syncthreads();  // aspT/o1 reads done; region A becomes pwl + o2sT
  if (wave < HEADS) {
    const int h = wave;
#pragma unroll
    for (int i = 0; i < 4; ++i)
#pragma unroll
      for (int r = 0; r < 8; ++r) {
        int t = i * 16 + r + mrow;            // spatial pixel
        o2sT[t * CH + h * 16 + ncol] = (_Float16)o2[i][r];
      }
  }
  for (int i = tid; i < (CH * CH) / 8; i += 256)
    ((v8h*)pwl)[i] = ((const v8h*)pw_h)[i];
  __syncthreads();

  // ---- Phase 8: proj 1x1 (96x64 = 96x96 @ 96x64) + window reverse --------
  for (int it = 0; it < 3; ++it) {
    int id = wave * 3 + it;                   // 0..23
    int mo = id >> 2, no = id & 3;
    v8f acc = {};
#pragma unroll
    for (int k0 = 0; k0 < CH; k0 += 32)
      acc = wmma32f16(load_a_rm32(pwl + mo * 16 * CH + k0, CH),
                      load_b_tr32(o2sT + (no * 16) * CH + k0, CH), acc);
#pragma unroll
    for (int r = 0; r < 8; ++r) {
      int o = mo * 16 + r + mrow;
      int t = no * 16 + ncol;
      out[o * (HH * WW) + (wy * 8 + (t >> 3)) * WW + wx * 8 + (t & 7)] =
          acc[r] + proj_b[o];
    }
  }
}

extern "C" void kernel_launch(void* const* d_in, const int* in_sizes, int n_in,
                              void* d_out, int out_size, void* d_ws, size_t ws_size,
                              hipStream_t stream) {
  const float* x           = (const float*)d_in[0];
  const float* qkv_w       = (const float*)d_in[1];
  const float* qkv_b       = (const float*)d_in[2];
  const float* dw_w        = (const float*)d_in[3];
  const float* dw_b        = (const float*)d_in[4];
  const float* proj_w      = (const float*)d_in[5];
  const float* proj_b      = (const float*)d_in[6];
  const float* temperature = (const float*)d_in[7];
  const float* rel_tab     = (const float*)d_in[8];
  float* out = (float*)d_out;

  char* ws = (char*)d_ws;                      // needs 73,728 B only
  _Float16* qw_h = (_Float16*)(ws);            // 288x96 f16
  _Float16* pw_h = (_Float16*)(ws + 55296);    // 96x96 f16

  prep_weights_f16<<<108, 256, 0, stream>>>(qkv_w, proj_w, qw_h, pw_h);
  fused_window_attn<<<4096, 256, 0, stream>>>(
      x, qw_h, qkv_b, dw_w, dw_b, pw_h, proj_b, temperature, rel_tab, out);
}
